// QNN_28157805593247
// MI455X (gfx1250) — matrix operand order repaired and checked by reading, not compiled, not run on hardware
//
#include <hip/hip_runtime.h>
#include <hip/hip_bf16.h>

typedef __attribute__((ext_vector_type(16))) _Float16 v16h;
typedef __attribute__((ext_vector_type(8)))  _Float16 v8h;
typedef __attribute__((ext_vector_type(8)))  float    v8f;
typedef __attribute__((ext_vector_type(4)))  float    f32x4;

#define T_STEPS 512
#define INP     24
#define HID     128
#define NG      512        // 4*H gate columns
#define AROW    168        // padded A row stride (halfs): 160 used, +8 pad (bank spread)
#define GSTR    20         // gates_lds row stride (floats): 16 used, +4 pad
#define BSL     16         // batch rows per workgroup (== WMMA M)
#define NTHREADS 512       // 16 waves (wave32)
#define NT_PER_WAVE 2      // 32 N-tiles / 16 waves

__device__ __forceinline__ float sigm(float x) {
    return __builtin_amdgcn_rcpf(1.0f + __expf(-x));        // v_exp + v_rcp
}
__device__ __forceinline__ float tanh_f(float x) {
    return 2.0f * __builtin_amdgcn_rcpf(1.0f + __expf(-2.0f * x)) - 1.0f;
}

__global__ __launch_bounds__(NTHREADS)
void lstm_wmma_kernel(const float* __restrict__ x,
                      const float* __restrict__ add_in,
                      const float* __restrict__ W_ih,
                      const float* __restrict__ W_hh,
                      const float* __restrict__ b_ih,
                      const float* __restrict__ b_hh,
                      const float* __restrict__ W1,
                      const float* __restrict__ b1,
                      const float* __restrict__ W2,
                      const float* __restrict__ b2,
                      float* __restrict__ out)
{
    // A staging: rows m=0..15, cols 0..127 = h_t (f16), 128..151 = x_t (f16), rest 0
    __shared__ __align__(16) _Float16 a_mat[BSL * AROW];       // 5.25 KB
    __shared__ __align__(16) float gates_lds[NG * GSTR];       // 40 KB, layout [n][m], padded
    __shared__ __align__(16) float h_lds[BSL * HID];           // 8 KB (final h only)
    __shared__ __align__(16) float bias_lds[NG];               // 2 KB
    __shared__ __align__(16) float z_lds[BSL * 64];            // 4 KB

    const int tid  = threadIdx.x;
    const int lane = tid & 31;
    const int wave = tid >> 5;
    const int b0   = blockIdx.x * BSL;
    const int mrow = lane & 15;   // A: row M; B: column offset; C/D: column N
    const int hi   = lane >> 4;   // lane-half selector per ISA VGPR layouts

    // ---------------- register-resident B fragments (weights, f16) ----------------
    // B[k][n] = Wcomb[n][k], Wcomb = [W_hh | W_ih | 0] shape [512][160].
    // B k order used here (A matches): lane holds column n = ntile*16 + lane%16,
    // 16 contiguous K values starting at k0 = c*32 + hi*16.
    v16h Bfrag[NT_PER_WAVE][5];
    #pragma unroll
    for (int nt = 0; nt < NT_PER_WAVE; ++nt) {
        const int n = (wave * NT_PER_WAVE + nt) * 16 + mrow;
        #pragma unroll
        for (int c = 0; c < 5; ++c) {
            const int k0 = c * 32 + hi * 16;
            v16h f;
            #pragma unroll
            for (int l = 0; l < 16; ++l) {
                const int k = k0 + l;
                float w = 0.0f;
                if (k < HID)              w = W_hh[n * HID + k];
                else if (k < HID + INP)   w = W_ih[n * INP + (k - HID)];
                f[l] = (_Float16)w;
            }
            Bfrag[nt][c] = f;
        }
    }

    // ---------------- init ----------------
    bias_lds[tid] = b_ih[tid] + b_hh[tid];
    #pragma unroll
    for (int r = 0; r < 6; ++r) {                 // zero all 16*168 = 2688 halfs
        const int idx = tid + NTHREADS * r;
        if (idx < BSL * AROW) a_mat[idx] = (_Float16)0.0f;
    }
    __syncthreads();
    // x staging indices: valid (duplicated) for ALL threads so the prefetch load
    // can be unconditional (no exec-masked region for the compiler to hoist into)
    const int  xi   = tid % (BSL * INP);
    const int  xm   = xi / INP, xk = xi % INP;
    const bool xthr = (tid < BSL * INP);
    const float* xbase = x + (size_t)(b0 + xm) * T_STEPS * INP + xk;
    if (xthr) a_mat[xm * AROW + HID + xk] = (_Float16)xbase[0];   // stage x_0
    __syncthreads();

    // loop-invariant bias for this thread's two N-tiles
    const float bias0 = bias_lds[(wave * NT_PER_WAVE + 0) * 16 + mrow];
    const float bias1 = bias_lds[(wave * NT_PER_WAVE + 1) * 16 + mrow];

    // cell state lives in registers: thread owns (m = tid&15, j = jb..jb+3) forever
    const int m2  = tid & 15;
    const int jb  = (tid >> 4) * 4;
    float creg[4] = {0.0f, 0.0f, 0.0f, 0.0f};

    // ---------------- recurrence over T ----------------
    #pragma unroll 1
    for (int t = 0; t < T_STEPS; ++t) {
        // unconditional early load of x_{t+1} (clamped); first USE is in phase 2,
        // so s_wait_loadcnt lands after the WMMA phase (latency hidden)
        const int tldr = (t + 1 < T_STEPS) ? (t + 1) : (T_STEPS - 1);
        const float xv = xbase[(size_t)tldr * INP];
        // pull x ~9 steps ahead into cache (global_prefetch_b8, no counter)
        const int tpf = (t + 9 < T_STEPS) ? (t + 9) : (T_STEPS - 1);
        __builtin_prefetch(&xbase[(size_t)tpf * INP], 0, 1);

        // ---- phase 1a: issue ALL A-fragment loads (10x ds_load_b128 in flight) ----
        v16h afr[5];
        #pragma unroll
        for (int c = 0; c < 5; ++c) {
            // A k order used here (matching B): lane (m, hi) holds K
            // {c*32+hi*16 .. +7} in VGPR0-3 and {c*32+hi*16+8 .. +7} in VGPR4-7.
            const v8h lo  = *(const v8h*)&a_mat[mrow * AROW + c * 32 + hi * 16];
            const v8h hi8 = *(const v8h*)&a_mat[mrow * AROW + c * 32 + hi * 16 + 8];
            #pragma unroll
            for (int l = 0; l < 8; ++l) { afr[c][l] = lo[l]; afr[c][l + 8] = hi8[l]; }
        }

        // ---- phase 1b: gates = [h | x] @ Wcomb^T + b  via WMMA ----
        v8f acc0, acc1;
        #pragma unroll
        for (int e = 0; e < 8; ++e) { acc0[e] = bias0; acc1[e] = bias1; }
        #pragma unroll
        for (int c = 0; c < 5; ++c) {
            acc0 = __builtin_amdgcn_wmma_f32_16x16x32_f16(
                false, afr[c], false, Bfrag[0][c], (short)0, acc0, false, false);
            acc1 = __builtin_amdgcn_wmma_f32_16x16x32_f16(
                false, afr[c], false, Bfrag[1][c], (short)0, acc1, false, false);
        }
        // C/D layout: VGPR r -> M = r + 8*hi, N = lane%16. Store gates as [n][m]:
        // two contiguous float4 stores per fragment.
        #pragma unroll
        for (int nt = 0; nt < NT_PER_WAVE; ++nt) {
            const int n = (wave * NT_PER_WAVE + nt) * 16 + mrow;
            float* gp = &gates_lds[n * GSTR + hi * 8];
            const v8f& a8 = (nt == 0) ? acc0 : acc1;
            f32x4 q0, q1;
            #pragma unroll
            for (int e = 0; e < 4; ++e) { q0[e] = a8[e]; q1[e] = a8[e + 4]; }
            *(f32x4*)gp       = q0;
            *(f32x4*)(gp + 4) = q1;
        }
        __syncthreads();

        // ---- phase 2a: load ALL 16 gate pre-activations (pipelined ds loads) ----
        float gv[4][4];
        #pragma unroll
        for (int r = 0; r < 4; ++r) {
            const int j = jb + r;
            gv[r][0] = gates_lds[(j            ) * GSTR + m2];
            gv[r][1] = gates_lds[(j +     HID) * GSTR + m2];
            gv[r][2] = gates_lds[(j + 2 * HID) * GSTR + m2];
            gv[r][3] = gates_lds[(j + 3 * HID) * GSTR + m2];
        }
        // ---- phase 2b: LSTM elementwise update (c in regs) + restage h/x ----
        #pragma unroll
        for (int r = 0; r < 4; ++r) {
            const int j = jb + r;
            const float ig = sigm  (gv[r][0]);
            const float fg = sigm  (gv[r][1]);
            const float gg = tanh_f(gv[r][2]);
            const float og = sigm  (gv[r][3]);
            const float cn = fg * creg[r] + ig * gg;
            creg[r] = cn;
            const float hn = og * tanh_f(cn);
            a_mat[m2 * AROW + j] = (_Float16)hn;
            if (t == T_STEPS - 1) h_lds[m2 * HID + j] = hn;   // uniform branch
        }
        if (xthr && (t + 1 < T_STEPS)) {
            a_mat[xm * AROW + HID + xk] = (_Float16)xv;       // cvt at store site
        }
        __syncthreads();
    }

    // ---------------- MLP head: relu([h, add]) @ W1^T + b1 -> relu -> @ W2^T + b2 ----
    #pragma unroll
    for (int r = 0; r < 2; ++r) {
        const int idx = tid * 2 + r;            // 0..1023 = 16 rows * 64 units
        const int m = idx >> 6, u = idx & 63;
        float s = b1[u];
        #pragma unroll 4
        for (int k = 0; k < HID; ++k)
            s += fmaxf(h_lds[m * HID + k], 0.0f) * W1[u * (HID + 2) + k];
        s += fmaxf(add_in[(b0 + m) * 2 + 0], 0.0f) * W1[u * (HID + 2) + HID];
        s += fmaxf(add_in[(b0 + m) * 2 + 1], 0.0f) * W1[u * (HID + 2) + HID + 1];
        z_lds[m * 64 + u] = s;
    }
    __syncthreads();
    if (tid < BSL * 3) {
        const int m = tid / 3, j = tid % 3;
        float s = b2[j];
        #pragma unroll 4
        for (int u = 0; u < 64; ++u)
            s += fmaxf(z_lds[m * 64 + u], 0.0f) * W2[j * 64 + u];
        out[(b0 + m) * 3 + j] = s;
    }
}

extern "C" void kernel_launch(void* const* d_in, const int* in_sizes, int n_in,
                              void* d_out, int out_size, void* d_ws, size_t ws_size,
                              hipStream_t stream) {
    const float* x      = (const float*)d_in[0];
    const float* add_in = (const float*)d_in[1];
    const float* W_ih   = (const float*)d_in[2];
    const float* W_hh   = (const float*)d_in[3];
    const float* b_ih   = (const float*)d_in[4];
    const float* b_hh   = (const float*)d_in[5];
    const float* W1     = (const float*)d_in[6];
    const float* b1     = (const float*)d_in[7];
    const float* W2     = (const float*)d_in[8];
    const float* b2     = (const float*)d_in[9];
    float* out = (float*)d_out;

    const int B = in_sizes[1] / 2;        // additional_input is [B, 2]
    const int grid = B / BSL;             // 64 persistent workgroups
    lstm_wmma_kernel<<<grid, NTHREADS, 0, stream>>>(
        x, add_in, W_ih, W_hh, b_ih, b_hh, W1, b1, W2, b2, out);
}
